// Descriptor_loss_39831526703619
// MI455X (gfx1250) — compile-verified, matrix operand order checked by reading, not run on hardware
//
#include <hip/hip_runtime.h>
#include <hip/hip_bf16.h>
#include <math.h>

typedef __attribute__((ext_vector_type(16))) _Float16 v16h;
typedef __attribute__((ext_vector_type(8)))  _Float16 v8h;
typedef __attribute__((ext_vector_type(8)))  float    v8f;

#define TFRAMES 20   // T*C = 5*4
#define BB      2
#define CF      32
#define HC      30
#define WC      40
#define MPIX    1200  // HC*WC, == 75*16
#define NTILES  75
#define IV0     1
#define IV1     4
#define NPAIR0  38   // (20-1)*2
#define NPAIRS  70   // 38 + (20-4)*2

// ---- helpers -------------------------------------------------------------

__device__ __forceinline__ void decode_pair(int pair, int& f0, int& f1, int& b, int& Niv) {
  if (pair < NPAIR0) {
    int n = pair >> 1; b = pair & 1; f0 = n; f1 = n + IV0; Niv = TFRAMES - IV0;
  } else {
    int p = pair - NPAIR0;
    int n = p >> 1; b = p & 1; f0 = n; f1 = n + IV1; Niv = TFRAMES - IV1;
  }
}

// Load a 16x32 f16 WMMA fragment from row-major (pixel, channel) data.
// Lane holds pixel (lane&15); lanes<16 take channels {0..7,16..23},
// lanes>=16 take {8..15,24..31}  (matches CDNA5 16-bit A/B VGPR layout).
__device__ __forceinline__ v16h load_frag(const _Float16* rowbase, int sel) {
  v8h lo = *reinterpret_cast<const v8h*>(rowbase + sel);
  v8h hi = *reinterpret_cast<const v8h*>(rowbase + sel + 16);
  v16h r;
  #pragma unroll
  for (int k = 0; k < 8; ++k) { r[k] = lo[k]; r[k + 8] = hi[k]; }
  return r;
}

__device__ __forceinline__ v8f wmma_tile(v16h a, v16h b) {
  v8f c = {};
  return __builtin_amdgcn_wmma_f32_16x16x32_f16(false, a, false, b, (short)0, c, false, false);
}

// ---- K_init: zero the output scalar --------------------------------------

__global__ void zero_kernel(float* out) { out[0] = 0.0f; }

// ---- K0: homography + per-pixel warp precompute --------------------------

__global__ void homog_warp_kernel(const float* __restrict__ rv, const float* __restrict__ tv,
                                  const float* __restrict__ nts, const float* __restrict__ dep,
                                  const float* __restrict__ Ks, const float* __restrict__ Kinv,
                                  const float* __restrict__ osz, float* __restrict__ warp) {
  __shared__ float Hs[9];
  int pair = blockIdx.x;
  int f0, f1, b, Niv;
  decode_pair(pair, f0, f1, b, Niv);
  if (threadIdx.x == 0) {
    // param index for flat-frame tc with batch b: (t*B + b)*C + c
    int t0 = f0 / 4, c0 = f0 % 4, t1 = f1 / 4, c1 = f1 % 4;
    int i0 = (t0 * BB + b) * 4 + c0;
    int i1 = (t1 * BB + b) * 4 + c1;

    auto rodr = [](const float* v, float* R) {
      float th = sqrtf(v[0]*v[0] + v[1]*v[1] + v[2]*v[2]);
      float inv = 1.0f / fmaxf(th, 1e-12f);
      float kx = v[0]*inv, ky = v[1]*inv, kz = v[2]*inv;
      float s = sinf(th), c = cosf(th), oc = 1.0f - c;
      R[0] = c + oc*kx*kx;    R[1] = oc*kx*ky - s*kz; R[2] = oc*kx*kz + s*ky;
      R[3] = oc*ky*kx + s*kz; R[4] = c + oc*ky*ky;    R[5] = oc*ky*kz - s*kx;
      R[6] = oc*kz*kx - s*ky; R[7] = oc*kz*ky + s*kx; R[8] = c + oc*kz*kz;
    };
    float R0[9], R1[9], R[9];
    rodr(rv + 3*i0, R0);
    rodr(rv + 3*i1, R1);
    // R = R1 * R0^T
    #pragma unroll
    for (int i = 0; i < 3; ++i)
      #pragma unroll
      for (int j = 0; j < 3; ++j)
        R[i*3+j] = R1[i*3+0]*R0[j*3+0] + R1[i*3+1]*R0[j*3+1] + R1[i*3+2]*R0[j*3+2];
    const float* tv0 = tv + 3*i0;
    const float* tv1 = tv + 3*i1;
    float tt[3];
    #pragma unroll
    for (int i = 0; i < 3; ++i)
      tt[i] = tv1[i] - (R[i*3+0]*tv0[0] + R[i*3+1]*tv0[1] + R[i*3+2]*tv0[2]);
    const float* nv = nts + 3*i0;
    float d = dep[i0];
    float Mv[9];
    #pragma unroll
    for (int i = 0; i < 3; ++i)
      #pragma unroll
      for (int j = 0; j < 3; ++j)
        Mv[i*3+j] = R[i*3+j] - tt[i]*nv[j]/d;
    const float* Km = Ks + 9*i0;
    const float* Ki = Kinv + 9*i0;
    float KM[9], Hm[9];
    #pragma unroll
    for (int i = 0; i < 3; ++i)
      #pragma unroll
      for (int j = 0; j < 3; ++j)
        KM[i*3+j] = Km[i*3+0]*Mv[0*3+j] + Km[i*3+1]*Mv[1*3+j] + Km[i*3+2]*Mv[2*3+j];
    #pragma unroll
    for (int i = 0; i < 3; ++i)
      #pragma unroll
      for (int j = 0; j < 3; ++j)
        Hm[i*3+j] = KM[i*3+0]*Ki[0*3+j] + KM[i*3+1]*Ki[1*3+j] + KM[i*3+2]*Ki[2*3+j];
    // scale homography: svec = [320/osz_w, 240/osz_h, 1]
    float sv[3];
    sv[0] = 320.0f / osz[2*i0 + 1];
    sv[1] = 240.0f / osz[2*i0 + 0];
    sv[2] = 1.0f;
    #pragma unroll
    for (int i = 0; i < 3; ++i)
      #pragma unroll
      for (int j = 0; j < 3; ++j)
        Hs[i*3+j] = Hm[i*3+j] * sv[i] / sv[j];
  }
  __syncthreads();
  for (int p = threadIdx.x; p < MPIX; p += blockDim.x) {
    int h = p / WC, w = p % WC;
    float x = (float)(w * 8), y = (float)(h * 8);
    float w0 = Hs[0]*x + Hs[1]*y + Hs[2];
    float w1 = Hs[3]*x + Hs[4]*y + Hs[5];
    float z  = Hs[6]*x + Hs[7]*y + Hs[8];
    z = (fabsf(z) < 1e-8f) ? 1e-8f : z;
    warp[((size_t)pair*MPIX + p)*2 + 0] = w0 / z;
    warp[((size_t)pair*MPIX + p)*2 + 1] = w1 / z;
  }
}

// ---- K1: channel-L2-normalize features, convert to f16 (pixel, channel) --

__global__ void normalize_kernel(const float* __restrict__ pred, _Float16* __restrict__ feat) {
  int tid = blockIdx.x * blockDim.x + threadIdx.x;
  if (tid >= TFRAMES * BB * MPIX) return;
  int fb = tid / MPIX;
  int p  = tid % MPIX;
  const float* base = pred + (size_t)fb * CF * MPIX + p;
  float v[CF];
  float ss = 0.0f;
  #pragma unroll
  for (int c = 0; c < CF; ++c) { v[c] = base[(size_t)c * MPIX]; ss += v[c]*v[c]; }
  float inv = 1.0f / fmaxf(sqrtf(ss), 1e-12f);
  _Float16* out = feat + ((size_t)fb * MPIX + p) * CF;
  #pragma unroll
  for (int c = 0; c < CF; ++c) out[c] = (_Float16)(v[c] * inv);
}

// ---- K2: row norms of relu(D0·D1^T) -------------------------------------

__global__ void rownorm_kernel(const _Float16* __restrict__ feat, float* __restrict__ rninv) {
  int i    = blockIdx.x;   // row tile
  int pair = blockIdx.y;
  int lane = threadIdx.x;
  int f0, f1, b, Niv;
  decode_pair(pair, f0, f1, b, Niv);
  int sel  = (lane >= 16) ? 8 : 0;
  int lrow = lane & 15;
  const _Float16* abase  = feat + ((size_t)(f0*BB + b)*MPIX + i*16 + lrow)*CF;
  const _Float16* bframe = feat + (size_t)(f1*BB + b)*MPIX*CF;
  v16h afrag = load_frag(abase, sel);
  float rs[8] = {0,0,0,0,0,0,0,0};
  for (int j = 0; j < NTILES; ++j) {
    v16h bfrag = load_frag(bframe + (size_t)(j*16 + lrow)*CF, sel);
    v8f acc = wmma_tile(afrag, bfrag);
    #pragma unroll
    for (int v = 0; v < 8; ++v) { float r = fmaxf(acc[v], 0.0f); rs[v] += r*r; }
  }
  #pragma unroll
  for (int v = 0; v < 8; ++v) {
    float s = rs[v];
    s += __shfl_xor(s, 1, 32); s += __shfl_xor(s, 2, 32);
    s += __shfl_xor(s, 4, 32); s += __shfl_xor(s, 8, 32);
    rs[v] = s;
  }
  if (lrow == 0) {
    int rbase = i*16 + ((lane >= 16) ? 8 : 0);
    #pragma unroll
    for (int v = 0; v < 8; ++v)
      rninv[(size_t)pair*MPIX + rbase + v] = 1.0f / fmaxf(sqrtf(rs[v]), 1e-12f);
  }
}

// ---- K3: column norms of row-normalized relu(dot) ------------------------

__global__ void colnorm_kernel(const _Float16* __restrict__ feat, const float* __restrict__ rninv,
                               float* __restrict__ cninv) {
  int j    = blockIdx.x;   // col tile
  int pair = blockIdx.y;
  int lane = threadIdx.x;
  int f0, f1, b, Niv;
  decode_pair(pair, f0, f1, b, Niv);
  int sel  = (lane >= 16) ? 8 : 0;
  int lrow = lane & 15;
  const _Float16* aframe = feat + (size_t)(f0*BB + b)*MPIX*CF;
  v16h bfrag = load_frag(feat + ((size_t)(f1*BB + b)*MPIX + j*16 + lrow)*CF, sel);
  float cs = 0.0f;
  for (int i = 0; i < NTILES; ++i) {
    v16h afrag = load_frag(aframe + (size_t)(i*16 + lrow)*CF, sel);
    v8f acc = wmma_tile(afrag, bfrag);
    const float* rn = rninv + (size_t)pair*MPIX + i*16 + ((lane >= 16) ? 8 : 0);
    #pragma unroll
    for (int v = 0; v < 8; ++v) {
      float r = fmaxf(acc[v], 0.0f) * rn[v];
      cs += r*r;
    }
  }
  cs += __shfl_xor(cs, 16, 32);
  if (lane < 16)
    cninv[(size_t)pair*MPIX + j*16 + lane] = 1.0f / fmaxf(sqrtf(cs), 1e-12f);
}

// ---- K4: masked margin loss + global reduction ---------------------------

__global__ void loss_kernel(const _Float16* __restrict__ feat, const float* __restrict__ rninv,
                            const float* __restrict__ cninv, const float* __restrict__ warp,
                            float* __restrict__ out) {
  int i    = blockIdx.x;   // row tile
  int pair = blockIdx.y;
  int lane = threadIdx.x;
  int f0, f1, b, Niv;
  decode_pair(pair, f0, f1, b, Niv);
  int sel  = (lane >= 16) ? 8 : 0;
  int lrow = lane & 15;
  const _Float16* abase  = feat + ((size_t)(f0*BB + b)*MPIX + i*16 + lrow)*CF;
  const _Float16* bframe = feat + (size_t)(f1*BB + b)*MPIX*CF;
  v16h afrag = load_frag(abase, sel);

  int rbase = i*16 + ((lane >= 16) ? 8 : 0);
  float irn[8], wx[8], wy[8];
  #pragma unroll
  for (int v = 0; v < 8; ++v) {
    int p = rbase + v;
    irn[v] = rninv[(size_t)pair*MPIX + p];
    wx[v]  = warp[((size_t)pair*MPIX + p)*2 + 0];
    wy[v]  = warp[((size_t)pair*MPIX + p)*2 + 1];
  }

  float part = 0.0f;
  for (int j = 0; j < NTILES; ++j) {
    v16h bfrag = load_frag(bframe + (size_t)(j*16 + lrow)*CF, sel);
    int q = j*16 + lrow;
    float cn = cninv[(size_t)pair*MPIX + q];
    float cx = (float)((q % WC) * 8);
    float cy = (float)((q / WC) * 8);
    v8f acc = wmma_tile(afrag, bfrag);
    #pragma unroll
    for (int v = 0; v < 8; ++v) {
      float val = fmaxf(acc[v], 0.0f) * irn[v] * cn;
      float pos = fmaxf(1.0f - val, 0.0f);
      float neg = fmaxf(val - 0.2f, 0.0f);
      float dx = wx[v] - cx, dy = wy[v] - cy;
      bool m = (dx*dx + dy*dy) <= 56.25f;   // (SCALE-0.5)^2 = 7.5^2
      part += m ? 0.05f * pos : neg;
    }
  }
  part += __shfl_xor(part, 1, 32);  part += __shfl_xor(part, 2, 32);
  part += __shfl_xor(part, 4, 32);  part += __shfl_xor(part, 8, 32);
  part += __shfl_xor(part, 16, 32);
  if (lane == 0) {
    // per-element weight: 1/(B*M*M) per step mean, /N per interval, /2 intervals
    float scale = 1.0f / (2.0f * (float)MPIX * (float)MPIX * (float)Niv * 2.0f);
    atomicAdd(out, part * scale);
  }
}

// ---- launch --------------------------------------------------------------

extern "C" void kernel_launch(void* const* d_in, const int* in_sizes, int n_in,
                              void* d_out, int out_size, void* d_ws, size_t ws_size,
                              hipStream_t stream) {
  const float* pred = (const float*)d_in[0];
  const float* rv   = (const float*)d_in[1];
  const float* tv   = (const float*)d_in[2];
  const float* nts  = (const float*)d_in[3];
  const float* dep  = (const float*)d_in[4];
  const float* Ks   = (const float*)d_in[5];
  const float* Kinv = (const float*)d_in[6];
  const float* osz  = (const float*)d_in[7];
  float* out = (float*)d_out;

  char* ws = (char*)d_ws;
  _Float16* feat = (_Float16*)ws;                                  // 20*2*1200*32*2  = 3,072,000 B
  float* warp    = (float*)(ws + 3072000);                         // 70*1200*2*4     =   672,000 B
  float* rninv   = (float*)(ws + 3072000 + 672000);                // 70*1200*4       =   336,000 B
  float* cninv   = (float*)(ws + 3072000 + 672000 + 336000);       // 70*1200*4       =   336,000 B

  zero_kernel<<<1, 1, 0, stream>>>(out);
  homog_warp_kernel<<<NPAIRS, 128, 0, stream>>>(rv, tv, nts, dep, Ks, Kinv, osz, warp);
  int npix = TFRAMES * BB * MPIX;
  normalize_kernel<<<(npix + 255) / 256, 256, 0, stream>>>(pred, feat);
  dim3 grid(NTILES, NPAIRS);
  rownorm_kernel<<<grid, 32, 0, stream>>>(feat, rninv);
  colnorm_kernel<<<grid, 32, 0, stream>>>(feat, rninv, cninv);
  loss_kernel<<<grid, 32, 0, stream>>>(feat, rninv, cninv, warp, out);
}
